// SelfAttentionLayer_21131239097185
// MI455X (gfx1250) — matrix operand order repaired and checked by reading, not compiled, not run on hardware
//
#include <hip/hip_runtime.h>
#include <stdint.h>

#define N_TOK 8192
#define DIM   256

#if defined(__has_builtin)
#  if __has_builtin(__builtin_amdgcn_tensor_load_to_lds)
#    define HAVE_TDM 1
#  endif
#endif
#ifndef HAVE_TDM
#  define HAVE_TDM 0
#endif

typedef _Float16 v8h  __attribute__((ext_vector_type(8)));
typedef _Float16 v16h __attribute__((ext_vector_type(16)));
typedef float    v8f  __attribute__((ext_vector_type(8)));
typedef unsigned int u32x4 __attribute__((ext_vector_type(4)));
typedef int          i32x4 __attribute__((ext_vector_type(4)));
typedef int          i32x8 __attribute__((ext_vector_type(8)));

static __device__ __forceinline__ v16h cat16(v8h a, v8h b) {
  return __builtin_shufflevector(a, b, 0,1,2,3,4,5,6,7,8,9,10,11,12,13,14,15);
}
static __device__ __forceinline__ v8h ldg8(const _Float16* p) {
  return *(const v8h*)p;
}
static __device__ __forceinline__ v8f wmma_f16(v16h a, v16h b, v8f c) {
  return __builtin_amdgcn_wmma_f32_16x16x32_f16(false, a, false, b, (short)0, c, false, false);
}
static __device__ __forceinline__ v16h ldB(const _Float16* __restrict__ buf,
                                           int row, int stride, int k0, int g) {
  const _Float16* p = buf + row * stride + k0 + 16 * g;
  return cat16(ldg8(p), ldg8(p + 8));
}

#if HAVE_TDM
// Issue a 2-D TDM load: tile_d1 rows x tile_d0 f16 elements, row stride stride0
// (elements) in global memory, with pad_amount(code)+1 DWORDs of LDS padding
// appended every 2^(pad_code+1) DWORDs (== one row).
static __device__ __forceinline__ void tdm_load_2d(
    unsigned lds_addr, const void* gptr,
    unsigned tensor_d0, unsigned tensor_d1, unsigned stride0,
    unsigned tile_d0, unsigned tile_d1,
    unsigned pad_interval_code, unsigned pad_amount_code) {
  unsigned long long ga = (unsigned long long)(uintptr_t)gptr;
  u32x4 g0;
  g0[0] = 1u;                                         // count=1, user descriptor
  g0[1] = lds_addr;                                   // LDS byte address
  g0[2] = (unsigned)ga;                               // global_addr[31:0]
  g0[3] = (unsigned)((ga >> 32) & 0x1FFFFFFu) | (2u << 30);  // addr[56:32] | type=2
  i32x8 g1;
  g1[0] = (int)((1u << 16)                            // data_size = 2 bytes
              | (1u << 20)                            // pad_enable
              | (pad_interval_code << 22)
              | (pad_amount_code << 25));             // workgroup_mask = 0
  g1[1] = (int)((tensor_d0 & 0xFFFFu) << 16);         // tensor_dim0[15:0]
  g1[2] = (int)((tensor_d0 >> 16) | ((tensor_d1 & 0xFFFFu) << 16));
  g1[3] = (int)((tensor_d1 >> 16) | (tile_d0 << 16)); // tile_dim0
  g1[4] = (int)tile_d1;                               // tile_dim1 (tile_dim2=0)
  g1[5] = (int)stride0;                               // tensor_dim0_stride[31:0]
  g1[6] = 0;
  g1[7] = 0;
  i32x4 gz = {0, 0, 0, 0};
#if __clang_major__ >= 23
  i32x8 gz8 = {0, 0, 0, 0, 0, 0, 0, 0};
  __builtin_amdgcn_tensor_load_to_lds(g0, g1, gz, gz, gz8, 0);
#else
  __builtin_amdgcn_tensor_load_to_lds(g0, g1, gz, gz, 0);
#endif
}
#endif

// ---------------- Kernel 0: W (f32, [d][j]) -> WT (f16, [j][d]) ----------------
__global__ __launch_bounds__(256) void wt_conv_kernel(
    const float* __restrict__ Wq, const float* __restrict__ Wk, const float* __restrict__ Wv,
    _Float16* __restrict__ WTq, _Float16* __restrict__ WTk, _Float16* __restrict__ WTv) {
  int j = blockIdx.x;
  int d = threadIdx.x;
  WTq[j*DIM + d] = (_Float16)Wq[d*DIM + j];
  WTk[j*DIM + d] = (_Float16)Wk[d*DIM + j];
  WTv[j*DIM + d] = (_Float16)Wv[d*DIM + j];
}

static __device__ __forceinline__ void proj_tiles(
    const v16h (&a)[8], const _Float16* __restrict__ WT,
    int wave, int nn, int g, v8f (&acc)[4]) {
  #pragma unroll
  for (int u = 0; u < 4; ++u) {
    const int jt = wave + 4 * u;
    v8f t = {};
    v16h b = ldB(WT, jt * 16 + nn, DIM, 0, g);
    #pragma unroll
    for (int c = 0; c < 8; ++c) {
      v16h nb = b;
      if (c < 7) nb = ldB(WT, jt * 16 + nn, DIM, 32 * (c + 1), g);
      t = wmma_f16(a[c], b, t);
      b = nb;
    }
    acc[u] = t;
  }
}

// ---------------- Kernel 1: QKV projection via WMMA ----------------
__global__ __launch_bounds__(128) void qkv_kernel(
    const float* __restrict__ x,
    const _Float16* __restrict__ WTq, const _Float16* __restrict__ WTk,
    const _Float16* __restrict__ WTv,
    _Float16* __restrict__ Qh, _Float16* __restrict__ Kh, _Float16* __restrict__ VT) {
  __shared__ __align__(16) _Float16 xs[16][264];
  const int tid = threadIdx.x;
  const int i0 = blockIdx.x * 16;

  for (int e = tid; e < 16 * DIM; e += 128) {
    int r = e >> 8, c = e & 255;
    xs[r][c] = (_Float16)x[(i0 + r) * DIM + c];
  }
  __syncthreads();

  const int wave = tid >> 5, lane = tid & 31;
  const int g = lane >> 4, nn = lane & 15;

  v16h a[8];
  #pragma unroll
  for (int c = 0; c < 8; ++c) {
    const _Float16* p = &xs[nn][32 * c + 16 * g];
    a[c] = cat16(*(const v8h*)p, *(const v8h*)(p + 8));
  }

  v8f accQ[4], accK[4], accV[4];
  proj_tiles(a, WTq, wave, nn, g, accQ);
  proj_tiles(a, WTk, wave, nn, g, accK);
  proj_tiles(a, WTv, wave, nn, g, accV);

  #pragma unroll
  for (int u = 0; u < 4; ++u) {
    const int jt = wave + 4 * u;
    #pragma unroll
    for (int r = 0; r < 8; ++r) {
      const int row = i0 + r + 8 * g;
      const int col = jt * 16 + nn;
      Qh[row * DIM + col] = (_Float16)accQ[u][r];
      Kh[row * DIM + col] = (_Float16)accK[u][r];
      VT[col * N_TOK + row] = (_Float16)accV[u][r];
    }
  }
}

// ---------------- Kernel 2: causal flash attention via WMMA + TDM staging ----------------
// Block = 128 threads (4 waves); each wave owns a 16-row query tile. All waves
// share the same key-tile sequence, so K/V tiles are staged once per block in
// LDS (TDM double-buffered where available).
__global__ __launch_bounds__(128) void attn_kernel(
    const _Float16* __restrict__ Qh, const _Float16* __restrict__ Kh,
    const _Float16* __restrict__ VT, float* __restrict__ out) {
  __shared__ __align__(16) _Float16 ks[2][32][264];   // K tile: 32 keys x 256 d (+16B row pad)
  __shared__ __align__(16) _Float16 vs[2][256][40];   // V^T tile: 256 d x 32 keys (+16B row pad)
  __shared__ __align__(16) _Float16 pbuf[4][16][40];  // per-wave P tile
  const int tid = threadIdx.x;
  const int wave = tid >> 5, lane = tid & 31;
  const int g = lane >> 4, nn = lane & 15;
  const int it = blockIdx.x * 4 + wave;
  const int i0 = it * 16;

  v16h q[8];
  #pragma unroll
  for (int c = 0; c < 8; ++c) {
    const _Float16* p = Qh + (i0 + nn) * DIM + 32 * c + 16 * g;
    q[c] = cat16(ldg8(p), ldg8(p + 8));
  }

  v8f zero = {};
  v8f O[16];
  #pragma unroll
  for (int t = 0; t < 16; ++t) O[t] = zero;
  float mrow[8], lrow[8];
  #pragma unroll
  for (int r = 0; r < 8; ++r) { mrow[r] = -3.0e38f; lrow[r] = 0.f; }

  // Uniform key range per block (safe barriers). Block 0 covers ALL keys to
  // reproduce the reference's fully-masked row 0 (-1e9 absorption -> uniform softmax).
  const int bx = blockIdx.x;
  const int kend = (bx == 0) ? (N_TOK / 32) : ((bx * 64 + 62) / 32 + 1);

#if HAVE_TDM
  // K tile: row = 256 f16 = 128 DW -> pad_interval code 6; pad 4 DW -> code 3 (528B stride)
  // V tile: row =  32 f16 =  16 DW -> pad_interval code 3; pad 4 DW -> code 3 (80B stride)
  if (wave == 0) {
    tdm_load_2d((unsigned)(uintptr_t)&ks[0][0][0], Kh, DIM, N_TOK, DIM, DIM, 32, 6, 3);
    tdm_load_2d((unsigned)(uintptr_t)&vs[0][0][0], VT, N_TOK, DIM, N_TOK, 32, DIM, 3, 3);
  }
#endif

  for (int kb = 0; kb < kend; ++kb) {
    const int kbase = kb * 32;
    const int cur = kb & 1;

#if HAVE_TDM
    if (wave == 0) {
      if (kb + 1 < kend) {
        const int nxt = cur ^ 1;
        tdm_load_2d((unsigned)(uintptr_t)&ks[nxt][0][0],
                    Kh + (size_t)(kbase + 32) * DIM, DIM, N_TOK, DIM, DIM, 32, 6, 3);
        tdm_load_2d((unsigned)(uintptr_t)&vs[nxt][0][0],
                    VT + (kbase + 32), N_TOK, DIM, N_TOK, 32, DIM, 3, 3);
        __builtin_amdgcn_s_wait_tensorcnt(2);   // retire the pair filling `cur`
      } else {
        __builtin_amdgcn_s_wait_tensorcnt(0);
      }
    }
    __syncthreads();  // staged tiles visible to all waves
#else
    __syncthreads();  // previous-iteration reads done
    for (int e = tid; e < 32 * 32; e += 128) {            // K tile: 32 rows x 32 v8h
      int r = e >> 5, cc = e & 31;
      *(v8h*)&ks[cur][r][cc * 8] = ldg8(Kh + (size_t)(kbase + r) * DIM + cc * 8);
    }
    for (int e = tid; e < 256 * 4; e += 128) {            // V tile: 256 rows x 4 v8h
      int r = e >> 2, cc = e & 3;
      *(v8h*)&vs[cur][r][cc * 8] = ldg8(VT + (size_t)r * N_TOK + kbase + cc * 8);
    }
    __syncthreads();
#endif

    // ---- S = Q @ K^T from LDS (two 16-col key sub-tiles) ----
    v8f s0 = zero, s1 = zero;
    #pragma unroll
    for (int c = 0; c < 8; ++c) {
      const _Float16* kp0 = &ks[cur][nn][32 * c + 16 * g];
      const _Float16* kp1 = &ks[cur][16 + nn][32 * c + 16 * g];
      s0 = wmma_f16(q[c], cat16(*(const v8h*)kp0, *(const v8h*)(kp0 + 8)), s0);
      s1 = wmma_f16(q[c], cat16(*(const v8h*)kp1, *(const v8h*)(kp1 + 8)), s1);
    }

    // ---- online softmax ----
    float corr[8];
    #pragma unroll
    for (int r = 0; r < 8; ++r) {
      const int rowi = i0 + r + 8 * g;
      float v0 = s0[r] * 0.0625f + (((kbase + nn)      >= rowi) ? -1.0e9f : 0.f);
      float v1 = s1[r] * 0.0625f + (((kbase + 16 + nn) >= rowi) ? -1.0e9f : 0.f);
      float tmax = fmaxf(v0, v1);
      tmax = fmaxf(tmax, __shfl_xor(tmax, 1, 32));
      tmax = fmaxf(tmax, __shfl_xor(tmax, 2, 32));
      tmax = fmaxf(tmax, __shfl_xor(tmax, 4, 32));
      tmax = fmaxf(tmax, __shfl_xor(tmax, 8, 32));
      float mnew = fmaxf(mrow[r], tmax);
      float cr = __expf(mrow[r] - mnew);
      float p0 = __expf(v0 - mnew);
      float p1 = __expf(v1 - mnew);
      float rs = p0 + p1;
      rs += __shfl_xor(rs, 1, 32);
      rs += __shfl_xor(rs, 2, 32);
      rs += __shfl_xor(rs, 4, 32);
      rs += __shfl_xor(rs, 8, 32);
      lrow[r] = lrow[r] * cr + rs;
      mrow[r] = mnew;
      corr[r] = cr;
      pbuf[wave][r + 8 * g][nn]      = (_Float16)p0;
      pbuf[wave][r + 8 * g][16 + nn] = (_Float16)p1;
    }
    #pragma unroll
    for (int t = 0; t < 16; ++t) {
      #pragma unroll
      for (int r = 0; r < 8; ++r) O[t][r] *= corr[r];
    }
    __syncthreads();  // order P stores before A-fragment reload

    const _Float16* pr = &pbuf[wave][nn][0];
    v16h pa = cat16(*(const v8h*)(pr + 16 * g), *(const v8h*)(pr + 8 + 16 * g));

    // ---- O += P @ V from LDS ----
    #pragma unroll
    for (int t = 0; t < 16; ++t) {
      const _Float16* vp = &vs[cur][t * 16 + nn][16 * g];
      O[t] = wmma_f16(pa, cat16(*(const v8h*)vp, *(const v8h*)(vp + 8)), O[t]);
    }
    __syncthreads();  // all reads of ks/vs/pbuf done before next stage/issue
  }

  #pragma unroll
  for (int r = 0; r < 8; ++r) {
    const float inv = 1.0f / lrow[r];
    const int row = i0 + r + 8 * g;
    #pragma unroll
    for (int t = 0; t < 16; ++t)
      out[row * DIM + t * 16 + nn] = O[t][r] * inv;
  }
}

extern "C" void kernel_launch(void* const* d_in, const int* in_sizes, int n_in,
                              void* d_out, int out_size, void* d_ws, size_t ws_size,
                              hipStream_t stream) {
  const float* x  = (const float*)d_in[0];
  const float* Wq = (const float*)d_in[1];
  const float* Wk = (const float*)d_in[2];
  const float* Wv = (const float*)d_in[3];

  char* ws = (char*)d_ws;
  _Float16* WTq = (_Float16*)(ws);
  _Float16* WTk = (_Float16*)(ws + (128 << 10));
  _Float16* WTv = (_Float16*)(ws + (256 << 10));
  _Float16* Qh  = (_Float16*)(ws + (384 << 10));
  _Float16* Kh  = Qh + (size_t)N_TOK * DIM;
  _Float16* VT  = Kh + (size_t)N_TOK * DIM;

  wt_conv_kernel<<<256, 256, 0, stream>>>(Wq, Wk, Wv, WTq, WTk, WTv);
  qkv_kernel<<<N_TOK / 16, 128, 0, stream>>>(x, WTq, WTk, WTv, Qh, Kh, VT);
  attn_kernel<<<N_TOK / 64, 128, 0, stream>>>(Qh, Kh, VT, (float*)d_out);
}
